// DecoderGAT_37915971289822
// MI455X (gfx1250) — matrix operand order repaired and checked by reading, not compiled, not run on hardware
//
#include <hip/hip_runtime.h>
#include <hip/hip_bf16.h>

// ---------------------------------------------------------------------------
// GATConv forward for MI455X (gfx1250, wave32, WMMA).
//   K0: init out=bias, nmax=0(=-inf key), denom=0
//   K1: xp = x@W via v_wmma_f32_16x16x32_bf16, fused a_src/a_dst epilogue
//   K2: segment-max over dst via ordered-uint atomicMax
//   K3: segment-sum of exp(e-max) via f32 atomicAdd
//   K4: out[dst] += alpha * xp[src] via f32 atomicAdd (8 per edge-head)
// ---------------------------------------------------------------------------

typedef __attribute__((ext_vector_type(16))) __bf16 v16bf;
typedef __attribute__((ext_vector_type(8)))  float  v8f;

union BF16x16 { v16bf v; __bf16 e[16]; };

#define NEG_SLOPE 0.2f

// Native f32 -> bf16 convert: let the backend pick v_cvt_pk_bf16_f32 (RNE)
// instead of a 3-VALU-op emulation sequence.
__device__ __forceinline__ __bf16 f2bf(float f) { return (__bf16)f; }

// Monotone float -> uint key so unsigned atomicMax orders like float max.
__device__ __forceinline__ unsigned fkey(float f) {
  unsigned u = __float_as_uint(f);
  return (u & 0x80000000u) ? ~u : (u | 0x80000000u);
}
__device__ __forceinline__ float fdec(unsigned k) {
  unsigned u = (k & 0x80000000u) ? (k & 0x7FFFFFFFu) : ~k;
  return __uint_as_float(u);
}

__device__ __forceinline__ void edge_nodes(const long long* __restrict__ ei,
                                           long long E, long long eidx,
                                           int& src, int& dst) {
  if (eidx < E) { src = (int)ei[eidx]; dst = (int)ei[E + eidx]; }
  else          { src = dst = (int)(eidx - E); }            // self loops
}

// ---------------------------------------------------------------------------
// K0: initialization (runs every call; d_out/d_ws are poisoned by harness)
// ---------------------------------------------------------------------------
__global__ __launch_bounds__(256)
void gat_init(float* __restrict__ out, const float* __restrict__ bias,
              unsigned* __restrict__ nmax, float* __restrict__ denom, int N) {
  long long stride = (long long)gridDim.x * 256;
  long long base   = (long long)blockIdx.x * 256 + threadIdx.x;
  long long total  = (long long)N * 32;
  for (long long i = base; i < total; i += stride) out[i] = bias[i & 31];
  long long t4 = (long long)N * 4;
  for (long long i = base; i < t4; i += stride) { nmax[i] = 0u; denom[i] = 0.f; }
}

// ---------------------------------------------------------------------------
// K1: projection GEMM (WMMA bf16 -> f32 acc) + attention-logit epilogue.
// Block = 256 threads = 8 waves; each wave owns a 16-row x 32-col tile.
// ---------------------------------------------------------------------------
__global__ __launch_bounds__(256)
void gat_proj_wmma(const float* __restrict__ x, const float* __restrict__ W,
                   const float* __restrict__ attS, const float* __restrict__ attD,
                   float* __restrict__ xp, float* __restrict__ asrc,
                   float* __restrict__ adst, int N) {
  __shared__ __bf16 WsT[32][136];   // bf16 W transposed [n][k], padded (bank-conflict free)
  __shared__ float  xps[128][33];   // f32 tile staging for epilogue

  const int tid = threadIdx.x;

  // Stage W (128x32 f32, 16KB) as transposed bf16 in LDS — one time per block.
  for (int i = tid; i < 128 * 32; i += 256) {
    int k = i >> 5, n = i & 31;
    WsT[n][k] = f2bf(W[i]);
  }
  __syncthreads();

  const int wave  = tid >> 5;
  const int lane  = tid & 31;
  const int m     = lane & 15;   // row-in-tile (A) / col-in-tile (B,C)
  const int hi    = lane >> 4;   // lane half select
  const int koffA = hi * 8;      // CDNA5 16-bit A layout: hi lanes hold K+8
  const int koffB = hi * 8;      // 16-bit B layout (same as A): hi lanes hold K+8

  int row  = blockIdx.x * 128 + wave * 16 + m;
  int rowc = row < N ? row : N - 1;                 // clamp; writes guarded later
  const float* xrow = x + (size_t)rowc * 128;

  // Hoist B fragments: W is constant across the whole K loop (64 VGPRs).
  BF16x16 b0[4], b1[4];
#pragma unroll
  for (int t = 0; t < 4; ++t) {
    const __bf16* p0 = &WsT[m][t * 32 + koffB];       // cols 0..15
    const __bf16* p1 = &WsT[m + 16][t * 32 + koffB];  // cols 16..31
#pragma unroll
    for (int j = 0; j < 16; ++j) { b0[t].e[j] = p0[j + (j & 8)]; b1[t].e[j] = p1[j + (j & 8)]; }
  }

  v8f acc0 = {}; v8f acc1 = {};
#pragma unroll
  for (int t = 0; t < 4; ++t) {
    const int kb = t * 32;
    // A fragment: elements j=0..7 -> K=kb+koffA+j ; j=8..15 -> K=kb+16+koffA+(j-8)
    BF16x16 a;
    const float4* pa = (const float4*)(xrow + kb + koffA);
    float4 f0 = pa[0], f1 = pa[1];
    const float4* pb = (const float4*)(xrow + kb + 16 + koffA);
    float4 f2 = pb[0], f3 = pb[1];
    a.e[0]  = f2bf(f0.x); a.e[1]  = f2bf(f0.y); a.e[2]  = f2bf(f0.z); a.e[3]  = f2bf(f0.w);
    a.e[4]  = f2bf(f1.x); a.e[5]  = f2bf(f1.y); a.e[6]  = f2bf(f1.z); a.e[7]  = f2bf(f1.w);
    a.e[8]  = f2bf(f2.x); a.e[9]  = f2bf(f2.y); a.e[10] = f2bf(f2.z); a.e[11] = f2bf(f2.w);
    a.e[12] = f2bf(f3.x); a.e[13] = f2bf(f3.y); a.e[14] = f2bf(f3.z); a.e[15] = f2bf(f3.w);

    acc0 = __builtin_amdgcn_wmma_f32_16x16x32_bf16(false, a.v, false, b0[t].v,
                                                   (short)0, acc0, false, false);
    acc1 = __builtin_amdgcn_wmma_f32_16x16x32_bf16(false, a.v, false, b1[t].v,
                                                   (short)0, acc1, false, false);
  }

  // C/D layout: VGPR r -> row r (lanes 0-15) / r+8 (lanes 16-31), col = lane%16.
#pragma unroll
  for (int r = 0; r < 8; ++r) {
    int mm = r + hi * 8;
    xps[wave * 16 + mm][m]      = acc0[r];
    xps[wave * 16 + mm][m + 16] = acc1[r];
  }
  __syncthreads();

  // Per-row epilogue: write xp row + fused attention dot products.
  if (tid < 128) {
    int grow = blockIdx.x * 128 + tid;
    if (grow < N) {
      float as[4] = {0, 0, 0, 0}, ad[4] = {0, 0, 0, 0};
      float v[32];
#pragma unroll
      for (int j = 0; j < 32; ++j) {
        float t = xps[tid][j];
        v[j] = t;
        as[j >> 3] += t * attS[j];
        ad[j >> 3] += t * attD[j];
      }
      float* op = xp + (size_t)grow * 32;
#pragma unroll
      for (int j = 0; j < 32; j += 4)
        *(float4*)(op + j) = make_float4(v[j], v[j + 1], v[j + 2], v[j + 3]);
#pragma unroll
      for (int h = 0; h < 4; ++h) {
        asrc[(size_t)grow * 4 + h] = as[h];
        adst[(size_t)grow * 4 + h] = ad[h];
      }
    }
  }
}

// ---------------------------------------------------------------------------
// K2: per-(edge,head) leaky-relu logit -> segment max over dst
// ---------------------------------------------------------------------------
__global__ __launch_bounds__(256)
void gat_edge_max(const long long* __restrict__ ei, const float* __restrict__ asrc,
                  const float* __restrict__ adst, unsigned* __restrict__ nmax,
                  int E, int N) {
  long long T = (long long)(E + N) * 4;
  long long gid = (long long)blockIdx.x * 256 + threadIdx.x;
  if (gid >= T) return;
  long long eidx = gid >> 2; int h = (int)(gid & 3);
  int src, dst; edge_nodes(ei, E, eidx, src, dst);
  float a = asrc[(size_t)src * 4 + h] + adst[(size_t)dst * 4 + h];
  float e = a > 0.f ? a : NEG_SLOPE * a;
  atomicMax(&nmax[(size_t)dst * 4 + h], fkey(e));
}

// ---------------------------------------------------------------------------
// K3: denom[dst,h] += exp(e - max)   (recompute e: cheaper than spilling 26MB)
// ---------------------------------------------------------------------------
__global__ __launch_bounds__(256)
void gat_edge_sum(const long long* __restrict__ ei, const float* __restrict__ asrc,
                  const float* __restrict__ adst, const unsigned* __restrict__ nmax,
                  float* __restrict__ denom, int E, int N) {
  long long T = (long long)(E + N) * 4;
  long long gid = (long long)blockIdx.x * 256 + threadIdx.x;
  if (gid >= T) return;
  long long eidx = gid >> 2; int h = (int)(gid & 3);
  int src, dst; edge_nodes(ei, E, eidx, src, dst);
  float a = asrc[(size_t)src * 4 + h] + adst[(size_t)dst * 4 + h];
  float e = a > 0.f ? a : NEG_SLOPE * a;
  float p = __expf(e - fdec(nmax[(size_t)dst * 4 + h]));
  __hip_atomic_fetch_add(&denom[(size_t)dst * 4 + h], p,
                         __ATOMIC_RELAXED, __HIP_MEMORY_SCOPE_AGENT);
}

// ---------------------------------------------------------------------------
// K4: out[dst, h*8+c] += alpha * xp[src, h*8+c]
// ---------------------------------------------------------------------------
__global__ __launch_bounds__(256)
void gat_edge_agg(const long long* __restrict__ ei, const float* __restrict__ asrc,
                  const float* __restrict__ adst, const unsigned* __restrict__ nmax,
                  const float* __restrict__ denom, const float* __restrict__ xp,
                  float* __restrict__ out, int E, int N) {
  long long T = (long long)(E + N) * 4;
  long long gid = (long long)blockIdx.x * 256 + threadIdx.x;
  if (gid >= T) return;
  long long eidx = gid >> 2; int h = (int)(gid & 3);
  int src, dst; edge_nodes(ei, E, eidx, src, dst);
  float a = asrc[(size_t)src * 4 + h] + adst[(size_t)dst * 4 + h];
  float e = a > 0.f ? a : NEG_SLOPE * a;
  float p = __expf(e - fdec(nmax[(size_t)dst * 4 + h]));
  float d = denom[(size_t)dst * 4 + h];
  float alpha = p / (d + 1e-16f);

  const float* xr = xp + (size_t)src * 32 + h * 8;
  float*       orow = out + (size_t)dst * 32 + h * 8;
  float4 m0 = *(const float4*)(xr);
  float4 m1 = *(const float4*)(xr + 4);
  float mv[8] = {m0.x, m0.y, m0.z, m0.w, m1.x, m1.y, m1.z, m1.w};
#pragma unroll
  for (int c = 0; c < 8; ++c)
    __hip_atomic_fetch_add(&orow[c], alpha * mv[c],
                           __ATOMIC_RELAXED, __HIP_MEMORY_SCOPE_AGENT);
}

// ---------------------------------------------------------------------------
extern "C" void kernel_launch(void* const* d_in, const int* in_sizes, int n_in,
                              void* d_out, int out_size, void* d_ws, size_t ws_size,
                              hipStream_t stream) {
  const float*     x    = (const float*)d_in[0];
  const long long* ei   = (const long long*)d_in[1];  // int64 per reference
  const float*     W    = (const float*)d_in[2];
  const float*     attS = (const float*)d_in[3];
  const float*     attD = (const float*)d_in[4];
  const float*     bias = (const float*)d_in[5];
  float*           out  = (float*)d_out;

  const int N = in_sizes[0] / 128;  // x is [N,128]
  const int E = in_sizes[1] / 2;    // edge_index is [2,E]

  // Workspace layout (N*48*4 bytes ~= 19.2 MB for N=100k)
  float*    xp    = (float*)d_ws;                       // [N,32]
  float*    asrc  = xp + (size_t)N * 32;                // [N,4]
  float*    adst  = asrc + (size_t)N * 4;               // [N,4]
  unsigned* nmax  = (unsigned*)(adst + (size_t)N * 4);  // [N,4] ordered keys
  float*    denom = (float*)(nmax + (size_t)N * 4);     // [N,4]

  gat_init<<<1024, 256, 0, stream>>>(out, bias, nmax, denom, N);
  gat_proj_wmma<<<(N + 127) / 128, 256, 0, stream>>>(x, W, attS, attD,
                                                     xp, asrc, adst, N);

  long long T = (long long)(E + N) * 4;
  unsigned eb = (unsigned)((T + 255) / 256);
  gat_edge_max<<<eb, 256, 0, stream>>>(ei, asrc, adst, nmax, E, N);
  gat_edge_sum<<<eb, 256, 0, stream>>>(ei, asrc, adst, nmax, denom, E, N);
  gat_edge_agg<<<eb, 256, 0, stream>>>(ei, asrc, adst, nmax, denom, xp, out, E, N);
}
